// EnrichedGATModule_3925600109370
// MI455X (gfx1250) — matrix operand order repaired and checked by reading, not compiled, not run on hardware
//
#include <hip/hip_runtime.h>

// ---------------- problem constants ----------------
#define NUSERS 50000
#define NITEMS 30000
#define NN     80000          // NUSERS + NITEMS
#define EE     640000
#define CC     64
#define DD     128
#define EXTRA  768
#define HH     4
#define LL     3
#define EDIM   2
#define K1     (DD + EXTRA)   // 896
#define N1     (2 * DD)       // 256
#define HD     (HH * DD)      // 512

typedef unsigned short ushort_t;
typedef __bf16 bf16_t;
typedef bf16_t v16bf __attribute__((ext_vector_type(16)));
typedef float  v8f   __attribute__((ext_vector_type(8)));
typedef unsigned int v4u __attribute__((ext_vector_type(4)));

__device__ __forceinline__ ushort_t f2bf(float f) {
    unsigned u = __float_as_uint(f);
    unsigned r = u + 0x7FFFu + ((u >> 16) & 1u);   // round-to-nearest-even
    return (ushort_t)(r >> 16);
}
__device__ __forceinline__ float leaky(float x) { return x > 0.f ? x : 0.2f * x; }

__device__ __forceinline__ void atomicMaxFloat(float* addr, float val) {
    unsigned* a = (unsigned*)addr;
    unsigned old = *a;
    while (__uint_as_float(old) < val) {
        unsigned assumed = old;
        old = atomicCAS(a, assumed, __float_as_uint(val));
        if (old == assumed) break;
    }
}

// ---------------- pack kernels (fp32 -> bf16) ----------------
__global__ void pack_bf16k(const float* __restrict__ in, ushort_t* __restrict__ out, long long n) {
    long long i = (long long)blockIdx.x * blockDim.x + threadIdx.x;
    if (i < n) out[i] = f2bf(in[i]);
}

// transposed pack: in [K,N] row-major -> out [N,K] (column-major weights for GEMM B)
__global__ void pack_bf16T(const float* __restrict__ in, ushort_t* __restrict__ out,
                           int K, int N) {
    long long i = (long long)blockIdx.x * blockDim.x + threadIdx.x;
    long long total = (long long)K * N;
    if (i >= total) return;
    int n = (int)(i / K), k = (int)(i % K);
    out[i] = f2bf(in[(long long)k * N + n]);
}

__global__ void pack_items(const float* __restrict__ emb, const float* __restrict__ extra,
                           ushort_t* __restrict__ out) {
    long long i = (long long)blockIdx.x * blockDim.x + threadIdx.x;
    long long total = (long long)NITEMS * K1;
    if (i >= total) return;
    int row = (int)(i / K1), col = (int)(i % K1);
    float v = (col < DD) ? emb[(long long)row * DD + col]
                         : extra[(long long)row * EXTRA + (col - DD)];
    out[i] = f2bf(v);
}

// ---------------- WMMA GEMM: C = act(A_bf16[M,K] @ BT_bf16[N,K]^T + bias) ----------------
// Block tile 64(M) x 128(N), K-step 32. 256 threads = 8 waves in a 4(m) x 2(n) grid;
// each wave owns a 16x64 output sub-tile = four v_wmma_f32_16x16x32_bf16 accumulators
// sharing one A fragment. A staged row-major, B pre-transposed in GLOBAL so staging
// and fragment loads are all aligned b128 ops.
#define TM 64
#define TN 128
#define TK 32
#define ASTR 40   // ushorts per A row (16B-aligned rows, padded)
#define BSTR 40   // ushorts per Bt column

__global__ __launch_bounds__(256) void wmma_gemm_bf16(
    const ushort_t* __restrict__ A, const ushort_t* __restrict__ BT,
    const float* __restrict__ bias, float* __restrict__ Cm,
    int M, int Ntot, int K, int relu)
{
    __shared__ ushort_t As[TM * ASTR];   // [row][k]
    __shared__ ushort_t Bt[TN * BSTR];   // [col][k]  (transposed)
    int tid = threadIdx.x;
    int lane = tid & 31, wid = tid >> 5;
    int wm = wid & 3, wn = wid >> 2;
    int bm = blockIdx.y * TM;
    int bn = blockIdx.x * TN;

    v8f acc0 = {}, acc1 = {}, acc2 = {}, acc3 = {};

    // per-thread staging coordinates
    int ar = tid >> 2;             // 0..63 A row
    int ac = (tid & 3) * 8;        // 0,8,16,24 A col (k)
    int bcs = tid >> 1;            // 0..127 B col
    int kcs = (tid & 1) * 16;      // 0 or 16 B k-base

    // wave fragment coordinates
    int arow = wm * 16 + (lane & 15);
    int kh = lane >> 4;                 // A k-half select
    int bcol = wn * 64 + (lane & 15);   // first of 4 fragment columns
    int kbw = (lane >> 4) << 4;         // B k-base (0 or 16)

    for (int k0 = 0; k0 < K; k0 += TK) {
        // stage A: one 16B chunk per thread (single uniform-per-thread tail guard)
        {
            v4u av = {};
            int gr = bm + ar;
            if (gr < M) av = *(const v4u*)(A + (long long)gr * K + k0 + ac);
            *(v4u*)&As[ar * ASTR + ac] = av;
        }
        // stage B: two aligned b128 loads from pre-transposed weights
        {
            const ushort_t* src = BT + (long long)(bn + bcs) * K + k0 + kcs;
            v4u b0 = *(const v4u*)(src);
            v4u b1 = *(const v4u*)(src + 8);
            *(v4u*)&Bt[bcs * BSTR + kcs]     = b0;
            *(v4u*)&Bt[bcs * BSTR + kcs + 8] = b1;
        }
        __syncthreads();

        // A fragment (ISA 16-bit A 16x32 layout): two contiguous 8-ushort runs
        union { v16bf v; v4u q[2]; } ua;
        ua.q[0] = *(const v4u*)&As[arow * ASTR + (kh << 3)];
        ua.q[1] = *(const v4u*)&As[arow * ASTR + 16 + (kh << 3)];

        // B fragments: same two 8-k runs as A at fixed column in Bt
        union { v16bf v; v4u q[2]; } ub;
        ub.q[0] = *(const v4u*)&Bt[(bcol)      * BSTR + (kh << 3)];
        ub.q[1] = *(const v4u*)&Bt[(bcol)      * BSTR + 16 + (kh << 3)];
        acc0 = __builtin_amdgcn_wmma_f32_16x16x32_bf16(false, ua.v, false, ub.v,
                                                       (short)0, acc0, false, false);
        ub.q[0] = *(const v4u*)&Bt[(bcol + 16) * BSTR + (kh << 3)];
        ub.q[1] = *(const v4u*)&Bt[(bcol + 16) * BSTR + 16 + (kh << 3)];
        acc1 = __builtin_amdgcn_wmma_f32_16x16x32_bf16(false, ua.v, false, ub.v,
                                                       (short)0, acc1, false, false);
        ub.q[0] = *(const v4u*)&Bt[(bcol + 32) * BSTR + (kh << 3)];
        ub.q[1] = *(const v4u*)&Bt[(bcol + 32) * BSTR + 16 + (kh << 3)];
        acc2 = __builtin_amdgcn_wmma_f32_16x16x32_bf16(false, ua.v, false, ub.v,
                                                       (short)0, acc2, false, false);
        ub.q[0] = *(const v4u*)&Bt[(bcol + 48) * BSTR + (kh << 3)];
        ub.q[1] = *(const v4u*)&Bt[(bcol + 48) * BSTR + 16 + (kh << 3)];
        acc3 = __builtin_amdgcn_wmma_f32_16x16x32_bf16(false, ua.v, false, ub.v,
                                                       (short)0, acc3, false, false);
        __syncthreads();
    }

    // C/D layout: VGPR r -> row = r + 8*(lane>>4), col = lane&15; bias hoisted (never null)
    int rbase = bm + wm * 16 + ((lane >> 4) << 3);
    int c0 = bn + wn * 64 + (lane & 15);
    float b0 = bias[c0], b1 = bias[c0 + 16], b2 = bias[c0 + 32], b3 = bias[c0 + 48];
    #pragma unroll
    for (int r = 0; r < 8; r++) {
        int row = rbase + r;
        if (row < M) {
            float v0 = acc0[r] + b0;
            float v1 = acc1[r] + b1;
            float v2 = acc2[r] + b2;
            float v3 = acc3[r] + b3;
            if (relu) {
                v0 = fmaxf(v0, 0.f); v1 = fmaxf(v1, 0.f);
                v2 = fmaxf(v2, 0.f); v3 = fmaxf(v3, 0.f);
            }
            long long rb = (long long)row * Ntot;
            Cm[rb + c0]      = v0;
            Cm[rb + c0 + 16] = v1;
            Cm[rb + c0 + 32] = v2;
            Cm[rb + c0 + 48] = v3;
        }
    }
}

// ---------------- GCN ----------------
__global__ void deg_init(float* deg, int n) {
    int i = blockIdx.x * blockDim.x + threadIdx.x;
    if (i < n) deg[i] = 1.f;
}
__global__ void deg_accum(const int* __restrict__ ei, const int* __restrict__ cl, float* deg) {
    int e = blockIdx.x * blockDim.x + threadIdx.x;
    if (e >= EE) return;
    int s = ei[e], d = ei[EE + e];
    if (cl[s] == cl[d]) atomicAdd(&deg[d], 1.f);
}
__global__ void gcn_scatter(const int* __restrict__ ei, const int* __restrict__ cl,
                            const float* __restrict__ deg, const float* __restrict__ XW,
                            float* __restrict__ Xc) {
    int e = blockIdx.x;
    int s = ei[e], d = ei[EE + e];
    if (cl[s] != cl[d]) return;
    float norm = rsqrtf(deg[s]) * rsqrtf(deg[d]);
    int t = threadIdx.x;
    atomicAdd(&Xc[(long long)d * DD + t], norm * XW[(long long)s * DD + t]);
}
__global__ void gcn_self(const float* __restrict__ XW, const float* __restrict__ deg,
                         const float* __restrict__ b, float* __restrict__ Xc) {
    int n = blockIdx.x, t = threadIdx.x;
    Xc[(long long)n * DD + t] += XW[(long long)n * DD + t] / deg[n] + b[t];
}

// ---------------- cluster pooling ----------------
__global__ void pool_accum(const float* __restrict__ Xc, const int* __restrict__ cl,
                           float* pooled, float* csize) {
    int n = blockIdx.x, t = threadIdx.x;
    int c = cl[n];
    atomicAdd(&pooled[c * DD + t], Xc[(long long)n * DD + t]);
    if (t == 0) atomicAdd(&csize[c], 1.f);
}
__global__ void pool_div(float* pooled, const float* csize) {
    int i = blockIdx.x * blockDim.x + threadIdx.x;
    if (i < CC * DD) pooled[i] /= fmaxf(csize[i / DD], 1.f);
}

// ---------------- cluster graph build ----------------
__global__ void cg_accum(const int* __restrict__ ei, const int* __restrict__ cl,
                         const float* __restrict__ ea, float* cnt, float* asum) {
    int e = blockIdx.x * blockDim.x + threadIdx.x;
    if (e >= EE) return;
    int cu = cl[ei[e]], cv = cl[ei[EE + e]];
    if (cu != cv) {
        int k = cu * CC + cv;
        atomicAdd(&cnt[k], 1.f);
        atomicAdd(&asum[k * 2],     ea[(long long)e * 2]);
        atomicAdd(&asum[k * 2 + 1], ea[(long long)e * 2 + 1]);
    }
}
__global__ void cg_avg(const float* cnt, const float* asum, float* avg) {
    int k = blockIdx.x * blockDim.x + threadIdx.x;
    if (k >= CC * CC) return;
    float c = fmaxf(cnt[k], 1.f);
    avg[k * 2]     = asum[k * 2] / c;
    avg[k * 2 + 1] = asum[k * 2 + 1] / c;
}
__global__ void cg_loop(const float* cnt, const float* avg, float* lattr) {
    int d = threadIdx.x;
    if (d >= CC) return;
    float s0 = 0.f, s1 = 0.f, ic = 0.f;
    for (int s = 0; s < CC; s++) {
        if (cnt[s * CC + d] > 0.f) {
            s0 += avg[(s * CC + d) * 2];
            s1 += avg[(s * CC + d) * 2 + 1];
            ic += 1.f;
        }
    }
    lattr[d * 2]     = s0 / fmaxf(ic, 1.f);
    lattr[d * 2 + 1] = s1 / fmaxf(ic, 1.f);
}

// ---------------- attention scores ----------------
__global__ void node_scores(const float* __restrict__ xh, const float* __restrict__ att,
                            float* __restrict__ out) {
    int n = blockIdx.x;
    int t = threadIdx.x;
    int h = t >> 5, lane = t & 31;
    const float* row = xh + (long long)n * HD + h * DD;
    const float* a = att + h * DD;
    float s = 0.f;
    for (int c = lane; c < DD; c += 32) s += row[c] * a[c];
    for (int off = 16; off; off >>= 1) s += __shfl_down(s, off, 32);
    if (lane == 0) out[n * HH + h] = s;
}

__global__ void edge_coef(const float* __restrict__ lin_edge, const float* __restrict__ att_edge,
                          float* __restrict__ wh) {
    int t = threadIdx.x;
    if (t >= EDIM * HH) return;
    int ed = t / HH, h = t % HH;
    float s = 0.f;
    for (int c = 0; c < DD; c++) s += lin_edge[ed * HD + h * DD + c] * att_edge[h * DD + c];
    wh[t] = s;
}

// ---------------- full-graph GAT edge kernels ----------------
__global__ void loop_attr_accum(const int* __restrict__ ei, const float* __restrict__ ea,
                                float* lsum, float* lcnt) {
    int e = blockIdx.x * blockDim.x + threadIdx.x;
    if (e >= EE) return;
    int d = ei[EE + e];
    atomicAdd(&lsum[d * 2],     ea[(long long)e * 2]);
    atomicAdd(&lsum[d * 2 + 1], ea[(long long)e * 2 + 1]);
    atomicAdd(&lcnt[d], 1.f);
}
__global__ void node_alpha_l(const float* __restrict__ a_s, const float* __restrict__ a_d,
                             const float* __restrict__ lsum, const float* __restrict__ lcnt,
                             const float* __restrict__ wh, float* alpha_l, float* m) {
    int i = blockIdx.x * blockDim.x + threadIdx.x;
    if (i >= NN * HH) return;
    int n = i / HH, h = i % HH;
    float c = fmaxf(lcnt[n], 1.f);
    float ale = (lsum[n * 2] / c) * wh[h] + (lsum[n * 2 + 1] / c) * wh[HH + h];
    float v = leaky(a_s[i] + a_d[i] + ale);
    alpha_l[i] = v;
    m[i] = v;
}
__global__ void edge_alpha_max(const int* __restrict__ ei, const float* __restrict__ ea,
                               const float* __restrict__ a_s, const float* __restrict__ a_d,
                               const float* __restrict__ wh, float* alpha_e, float* m) {
    int e = blockIdx.x * blockDim.x + threadIdx.x;
    if (e >= EE) return;
    int s = ei[e], d = ei[EE + e];
    float e0 = ea[(long long)e * 2], e1 = ea[(long long)e * 2 + 1];
    #pragma unroll
    for (int h = 0; h < HH; h++) {
        float v = leaky(a_s[s * HH + h] + a_d[d * HH + h] + e0 * wh[h] + e1 * wh[HH + h]);
        alpha_e[(long long)e * HH + h] = v;
        atomicMaxFloat(&m[d * HH + h], v);
    }
}
__global__ void node_p_init(const float* __restrict__ alpha_l, const float* __restrict__ m,
                            const float* __restrict__ xh, float* denom, float* __restrict__ num) {
    int n = blockIdx.x, t = threadIdx.x;
    __shared__ float pl[HH];
    if (t < HH) {
        float p = __expf(alpha_l[n * HH + t] - m[n * HH + t]);
        pl[t] = p;
        denom[n * HH + t] = p;
    }
    __syncthreads();
    #pragma unroll
    for (int h = 0; h < HH; h++)
        num[(long long)n * HD + h * DD + t] = pl[h] * xh[(long long)n * HD + h * DD + t];
}
__global__ void edge_p(const int* __restrict__ ei, const float* __restrict__ alpha_e,
                       const float* __restrict__ m, float* p_e, float* denom) {
    int e = blockIdx.x * blockDim.x + threadIdx.x;
    if (e >= EE) return;
    int d = ei[EE + e];
    #pragma unroll
    for (int h = 0; h < HH; h++) {
        float p = __expf(alpha_e[(long long)e * HH + h] - m[d * HH + h]);
        p_e[(long long)e * HH + h] = p;
        atomicAdd(&denom[d * HH + h], p);
    }
}
__global__ void edge_scatter(const int* __restrict__ ei, const float* __restrict__ p_e,
                             const float* __restrict__ xh, float* __restrict__ num) {
    int e = blockIdx.x, t = threadIdx.x;
    int s = ei[e], d = ei[EE + e];
    #pragma unroll
    for (int h = 0; h < HH; h++)
        atomicAdd(&num[(long long)d * HD + h * DD + t],
                  p_e[(long long)e * HH + h] * xh[(long long)s * HD + h * DD + t]);
}
__global__ void gat_final(const float* __restrict__ num, const float* __restrict__ denom,
                          const float* __restrict__ bias, float* __restrict__ Y) {
    int n = blockIdx.x, t = threadIdx.x;
    float s = 0.f;
    #pragma unroll
    for (int h = 0; h < HH; h++)
        s += num[(long long)n * HD + h * DD + t] / denom[n * HH + h];
    Y[(long long)n * DD + t] = 0.25f * s + bias[t];
}

// ---------------- cluster GAT ----------------
__global__ void cg_alpha(const float* __restrict__ asc, const float* __restrict__ adc,
                         const float* __restrict__ avg, const float* __restrict__ cnt,
                         const float* __restrict__ lattr, const float* __restrict__ wh,
                         float* logits) {
    int i = blockIdx.x * blockDim.x + threadIdx.x;
    if (i >= CC * CC * HH) return;
    int s = i / (CC * HH), r = i % (CC * HH), d = r / HH, h = r % HH;
    float v;
    if (s == d) {
        float ale = lattr[d * 2] * wh[h] + lattr[d * 2 + 1] * wh[HH + h];
        v = leaky(asc[d * HH + h] + adc[d * HH + h] + ale);
    } else if (cnt[s * CC + d] > 0.f) {
        float ae = avg[(s * CC + d) * 2] * wh[h] + avg[(s * CC + d) * 2 + 1] * wh[HH + h];
        v = leaky(asc[s * HH + h] + adc[d * HH + h] + ae);
    } else {
        v = -__builtin_inff();
    }
    logits[(s * CC + d) * HH + h] = v;
}
__global__ void cg_softmax(float* logits) {   // softmax over sources s, per (d,h)
    int t = blockIdx.x * blockDim.x + threadIdx.x;
    if (t >= CC * HH) return;
    int d = t / HH, h = t % HH;
    float mx = -__builtin_inff();
    for (int s = 0; s < CC; s++) mx = fmaxf(mx, logits[(s * CC + d) * HH + h]);
    float sum = 0.f;
    for (int s = 0; s < CC; s++) {
        float p = __expf(logits[(s * CC + d) * HH + h] - mx);
        logits[(s * CC + d) * HH + h] = p;
        sum += p;
    }
    for (int s = 0; s < CC; s++) logits[(s * CC + d) * HH + h] /= sum;
}
__global__ void cg_out(const float* __restrict__ p, const float* __restrict__ xhc,
                       const float* __restrict__ bias, float* __restrict__ clout) {
    int d = blockIdx.x, t = threadIdx.x;
    float s = 0.f;
    for (int src = 0; src < CC; src++)
        #pragma unroll
        for (int h = 0; h < HH; h++)
            s += p[(src * CC + d) * HH + h] * xhc[src * HD + h * DD + t];
    clout[d * DD + t] = 0.25f * s + bias[t];
}
__global__ void combine(const float* __restrict__ Xc, const float* __restrict__ clout,
                        const int* __restrict__ cl, float* __restrict__ Xcomb) {
    int n = blockIdx.x, t = threadIdx.x;
    Xcomb[(long long)n * DD + t] = Xc[(long long)n * DD + t] + clout[cl[n] * DD + t];
}

// ---------------- GraphNorm + residual ELU ----------------
__global__ void col_accum(const float* __restrict__ Y, const float* __restrict__ ms,
                          const float* __restrict__ meanbuf, float* out, int n, int mode) {
    int t = threadIdx.x;
    int r0 = blockIdx.x * 128;
    int rend = r0 + 128 < n ? r0 + 128 : n;
    float sub = mode ? ms[t] * (meanbuf[t] / (float)n) : 0.f;
    float s = 0.f;
    for (int r = r0; r < rend; r++) {
        float v = Y[(long long)r * DD + t] - sub;
        s += mode ? v * v : v;
    }
    atomicAdd(&out[t], s);
}
__global__ void norm_elu(const float* __restrict__ Y, const float* __restrict__ Xin,
                         const float* __restrict__ colsum, const float* __restrict__ colsq,
                         const float* __restrict__ w, const float* __restrict__ b,
                         const float* __restrict__ ms, float* __restrict__ out) {
    long long i = (long long)blockIdx.x * blockDim.x + threadIdx.x;
    if (i >= (long long)NN * DD) return;
    int t = (int)(i & (DD - 1));
    float mean = colsum[t] / (float)NN;
    float var  = colsq[t]  / (float)NN;
    float cv = Y[i] - ms[t] * mean;
    float v = w[t] * cv * rsqrtf(var + 1e-5f) + b[t] + Xin[i];
    out[i] = v > 0.f ? v : (__expf(v) - 1.f);
}

// ---------------- host launcher ----------------
static inline int cdiv(long long a, int b) { return (int)((a + b - 1) / b); }

extern "C" void kernel_launch(void* const* d_in, const int* in_sizes, int n_in,
                              void* d_out, int out_size, void* d_ws, size_t ws_size,
                              hipStream_t stream) {
    const float* extra    = (const float*)d_in[0];
    const int*   ei       = (const int*)  d_in[1];
    const float* eattr    = (const float*)d_in[2];
    const int*   cl       = (const int*)  d_in[3];
    const float* user_emb = (const float*)d_in[4];
    const float* item_emb = (const float*)d_in[5];
    const float* W1 = (const float*)d_in[6],  *b1 = (const float*)d_in[7];
    const float* W2 = (const float*)d_in[8],  *b2 = (const float*)d_in[9];
    const float* gcnW = (const float*)d_in[10], *gcnb = (const float*)d_in[11];
    const float* g_lin = (const float*)d_in[12], *g_lin_edge = (const float*)d_in[13];
    const float* g_att_src = (const float*)d_in[14], *g_att_dst = (const float*)d_in[15];
    const float* g_att_edge = (const float*)d_in[16], *g_bias = (const float*)d_in[17];
    const float* gat_lin = (const float*)d_in[18], *gat_lin_edge = (const float*)d_in[19];
    const float* gat_att_src = (const float*)d_in[20], *gat_att_dst = (const float*)d_in[21];
    const float* gat_att_edge = (const float*)d_in[22], *gat_bias = (const float*)d_in[23];
    const float* norm_w = (const float*)d_in[24], *norm_b = (const float*)d_in[25];
    const float* norm_ms = (const float*)d_in[26];
    float* out = (float*)d_out;

    // bump allocator over d_ws
    char* wsp = (char*)d_ws;
    auto alloc = [&](size_t bytes) -> void* {
        void* p = (void*)wsp;
        wsp += (bytes + 255) & ~(size_t)255;
        return p;
    };
    ushort_t* bfbuf    = (ushort_t*)alloc((size_t)NITEMS * K1 * 2);       // reused pack target
    ushort_t* w1bf     = (ushort_t*)alloc((size_t)K1 * N1 * 2);           // [N1][K1] transposed
    ushort_t* w2bf     = (ushort_t*)alloc((size_t)N1 * DD * 2);           // [DD][N1]
    ushort_t* gcnwbf   = (ushort_t*)alloc((size_t)DD * DD * 2);           // [DD][DD]
    ushort_t* glinbf   = (ushort_t*)alloc((size_t)DD * HD * 2);           // [HD][DD]
    ushort_t* gatlinbf = (ushort_t*)alloc((size_t)LL * DD * HD * 2);      // per-layer [HD][DD]
    float* zbias  = (float*)alloc((size_t)HD * 4);                        // zero bias
    float* h1     = (float*)alloc((size_t)NITEMS * N1 * 4);
    float* X      = (float*)alloc((size_t)NN * DD * 4);
    float* XW     = (float*)alloc((size_t)NN * DD * 4);
    float* Xc     = (float*)alloc((size_t)NN * DD * 4);
    float* Xcomb  = (float*)alloc((size_t)NN * DD * 4);
    float* Ybuf   = (float*)alloc((size_t)NN * DD * 4);
    float* xh     = (float*)alloc((size_t)NN * HD * 4);
    float* num    = (float*)alloc((size_t)NN * HD * 4);
    float* a_s    = (float*)alloc((size_t)NN * HH * 4);
    float* a_d    = (float*)alloc((size_t)NN * HH * 4);
    float* alphal = (float*)alloc((size_t)NN * HH * 4);
    float* mbuf   = (float*)alloc((size_t)NN * HH * 4);
    float* denom  = (float*)alloc((size_t)NN * HH * 4);
    float* alphae = (float*)alloc((size_t)EE * HH * 4);
    float* p_e    = (float*)alloc((size_t)EE * HH * 4);
    float* lsum   = (float*)alloc((size_t)NN * 2 * 4);
    float* lcnt   = (float*)alloc((size_t)NN * 4);
    float* deg    = (float*)alloc((size_t)NN * 4);
    float* pooled = (float*)alloc((size_t)CC * DD * 4);
    float* csize  = (float*)alloc((size_t)CC * 4);
    float* cnt    = (float*)alloc((size_t)CC * CC * 4);
    float* asum   = (float*)alloc((size_t)CC * CC * 2 * 4);
    float* avg    = (float*)alloc((size_t)CC * CC * 2 * 4);
    float* lattr  = (float*)alloc((size_t)CC * 2 * 4);
    float* xhc    = (float*)alloc((size_t)CC * HD * 4);
    float* asc    = (float*)alloc((size_t)CC * HH * 4);
    float* adc    = (float*)alloc((size_t)CC * HH * 4);
    float* logits = (float*)alloc((size_t)CC * CC * HH * 4);
    float* clout  = (float*)alloc((size_t)CC * DD * 4);
    float* wh     = (float*)alloc(8 * 4);
    float* colsum = (float*)alloc(DD * 4);
    float* colsq  = (float*)alloc(DD * 4);

    const int blocksE = cdiv(EE, 256);

    hipMemsetAsync(zbias, 0, (size_t)HD * 4, stream);

    // ---- pack weights to bf16, transposed to [N][K] ----
    pack_bf16T<<<cdiv((long long)K1 * N1, 256), 256, 0, stream>>>(W1, w1bf, K1, N1);
    pack_bf16T<<<cdiv((long long)N1 * DD, 256), 256, 0, stream>>>(W2, w2bf, N1, DD);
    pack_bf16T<<<cdiv((long long)DD * DD, 256), 256, 0, stream>>>(gcnW, gcnwbf, DD, DD);
    pack_bf16T<<<cdiv((long long)DD * HD, 256), 256, 0, stream>>>(g_lin, glinbf, DD, HD);
    for (int l = 0; l < LL; l++)
        pack_bf16T<<<cdiv((long long)DD * HD, 256), 256, 0, stream>>>(
            gat_lin + (size_t)l * DD * HD, gatlinbf + (size_t)l * DD * HD, DD, HD);

    // ---- item MLP: relu(concat @ W1 + b1) @ W2 + b2 ----
    pack_items<<<cdiv((long long)NITEMS * K1, 256), 256, 0, stream>>>(item_emb, extra, bfbuf);
    wmma_gemm_bf16<<<dim3(N1 / TN, cdiv(NITEMS, 64)), 256, 0, stream>>>(
        bfbuf, w1bf, b1, h1, NITEMS, N1, K1, 1);
    pack_bf16k<<<cdiv((long long)NITEMS * N1, 256), 256, 0, stream>>>(h1, bfbuf,
                                                                      (long long)NITEMS * N1);
    wmma_gemm_bf16<<<dim3(DD / TN, cdiv(NITEMS, 64)), 256, 0, stream>>>(
        bfbuf, w2bf, b2, X + (size_t)NUSERS * DD, NITEMS, DD, N1, 0);
    hipMemcpyAsync(X, user_emb, (size_t)NUSERS * DD * 4, hipMemcpyDeviceToDevice, stream);

    // ---- cluster GCN ----
    pack_bf16k<<<cdiv((long long)NN * DD, 256), 256, 0, stream>>>(X, bfbuf, (long long)NN * DD);
    wmma_gemm_bf16<<<dim3(DD / TN, cdiv(NN, 64)), 256, 0, stream>>>(
        bfbuf, gcnwbf, zbias, XW, NN, DD, DD, 0);
    deg_init<<<cdiv(NN, 256), 256, 0, stream>>>(deg, NN);
    deg_accum<<<blocksE, 256, 0, stream>>>(ei, cl, deg);
    hipMemsetAsync(Xc, 0, (size_t)NN * DD * 4, stream);
    gcn_scatter<<<EE, DD, 0, stream>>>(ei, cl, deg, XW, Xc);
    gcn_self<<<NN, DD, 0, stream>>>(XW, deg, gcnb, Xc);

    // ---- pooling per cluster ----
    hipMemsetAsync(pooled, 0, (size_t)CC * DD * 4, stream);
    hipMemsetAsync(csize, 0, (size_t)CC * 4, stream);
    pool_accum<<<NN, DD, 0, stream>>>(Xc, cl, pooled, csize);
    pool_div<<<cdiv(CC * DD, 256), 256, 0, stream>>>(pooled, csize);

    // ---- cluster graph + cluster GAT ----
    hipMemsetAsync(cnt, 0, (size_t)CC * CC * 4, stream);
    hipMemsetAsync(asum, 0, (size_t)CC * CC * 2 * 4, stream);
    cg_accum<<<blocksE, 256, 0, stream>>>(ei, cl, eattr, cnt, asum);
    cg_avg<<<cdiv(CC * CC, 256), 256, 0, stream>>>(cnt, asum, avg);
    cg_loop<<<1, 64, 0, stream>>>(cnt, avg, lattr);
    pack_bf16k<<<cdiv((long long)CC * DD, 256), 256, 0, stream>>>(pooled, bfbuf,
                                                                  (long long)CC * DD);
    wmma_gemm_bf16<<<dim3(HD / TN, 1), 256, 0, stream>>>(bfbuf, glinbf, zbias, xhc,
                                                         CC, HD, DD, 0);
    node_scores<<<CC, 128, 0, stream>>>(xhc, g_att_src, asc);
    node_scores<<<CC, 128, 0, stream>>>(xhc, g_att_dst, adc);
    edge_coef<<<1, 64, 0, stream>>>(g_lin_edge, g_att_edge, wh);
    cg_alpha<<<cdiv(CC * CC * HH, 256), 256, 0, stream>>>(asc, adc, avg, cnt, lattr, wh, logits);
    cg_softmax<<<1, 256, 0, stream>>>(logits);
    cg_out<<<CC, DD, 0, stream>>>(logits, xhc, g_bias, clout);
    combine<<<NN, DD, 0, stream>>>(Xc, clout, cl, Xcomb);

    // ---- 3 GAT layers ----
    for (int l = 0; l < LL; l++) {
        const float* att_s = gat_att_src + (size_t)l * HH * DD;
        const float* att_d = gat_att_dst + (size_t)l * HH * DD;
        const float* att_e = gat_att_edge + (size_t)l * HH * DD;
        const float* lin_e = gat_lin_edge + (size_t)l * EDIM * HD;
        const float* bias_l = gat_bias + (size_t)l * DD;

        pack_bf16k<<<cdiv((long long)NN * DD, 256), 256, 0, stream>>>(Xcomb, bfbuf,
                                                                      (long long)NN * DD);
        wmma_gemm_bf16<<<dim3(HD / TN, cdiv(NN, 64)), 256, 0, stream>>>(
            bfbuf, gatlinbf + (size_t)l * DD * HD, zbias, xh, NN, HD, DD, 0);

        node_scores<<<NN, 128, 0, stream>>>(xh, att_s, a_s);
        node_scores<<<NN, 128, 0, stream>>>(xh, att_d, a_d);
        edge_coef<<<1, 64, 0, stream>>>(lin_e, att_e, wh);

        hipMemsetAsync(lsum, 0, (size_t)NN * 2 * 4, stream);
        hipMemsetAsync(lcnt, 0, (size_t)NN * 4, stream);
        loop_attr_accum<<<blocksE, 256, 0, stream>>>(ei, eattr, lsum, lcnt);
        node_alpha_l<<<cdiv((long long)NN * HH, 256), 256, 0, stream>>>(a_s, a_d, lsum, lcnt,
                                                                        wh, alphal, mbuf);
        edge_alpha_max<<<blocksE, 256, 0, stream>>>(ei, eattr, a_s, a_d, wh, alphae, mbuf);
        node_p_init<<<NN, DD, 0, stream>>>(alphal, mbuf, xh, denom, num);
        edge_p<<<blocksE, 256, 0, stream>>>(ei, alphae, mbuf, p_e, denom);
        edge_scatter<<<EE, DD, 0, stream>>>(ei, p_e, xh, num);
        gat_final<<<NN, DD, 0, stream>>>(num, denom, bias_l, Ybuf);

        hipMemsetAsync(colsum, 0, DD * 4, stream);
        hipMemsetAsync(colsq, 0, DD * 4, stream);
        col_accum<<<cdiv(NN, 128), DD, 0, stream>>>(Ybuf, norm_ms + (size_t)l * DD,
                                                    colsum, colsum, NN, 0);
        col_accum<<<cdiv(NN, 128), DD, 0, stream>>>(Ybuf, norm_ms + (size_t)l * DD,
                                                    colsum, colsq, NN, 1);
        float* dst = (l == LL - 1) ? out : Xcomb;
        norm_elu<<<cdiv((long long)NN * DD, 256), 256, 0, stream>>>(
            Ybuf, Xcomb, colsum, colsq, norm_w + (size_t)l * DD, norm_b + (size_t)l * DD,
            norm_ms + (size_t)l * DD, dst);
    }
}
